// Attention_22857815949599
// MI455X (gfx1250) — matrix operand order repaired and checked by reading, not compiled, not run on hardware
//
#include <hip/hip_runtime.h>
#include <hip/hip_bf16.h>

typedef __bf16 bf16;
typedef __attribute__((ext_vector_type(8)))  bf16  v8bf;
typedef __attribute__((ext_vector_type(16))) bf16  v16bf;
typedef __attribute__((ext_vector_type(8)))  float v8f;

constexpr int BB = 16;      // batch
constexpr int SS = 2048;    // seq len
constexpr int EE = 128;     // n_embd
constexpr int HH = 128;     // head dim
constexpr float SCALE = 0.08838834764831845f;  // 1/sqrt(128)

__device__ __forceinline__ v16bf cat8(v8bf lo, v8bf hi) {
  v16bf r;
#pragma unroll
  for (int i = 0; i < 8; ++i) { r[i] = lo[i]; r[8 + i] = hi[i]; }
  return r;
}

// ---------------------------------------------------------------------------
// fp32 -> bf16 weight preconversion (once per launch; W is loop-invariant
// across all 2048 projection tiles, so convert it exactly once).
// ---------------------------------------------------------------------------
__global__ __launch_bounds__(256) void cvt_bf16_kernel(
    const float* __restrict__ src, bf16* __restrict__ dst, int n) {
  const int i = blockIdx.x * 256 + threadIdx.x;
  if (i < n) dst[i] = (bf16)src[i];
}

// ---------------------------------------------------------------------------
// Projection: Y[16x128 tile] = X[16x128] @ W^T + b, bf16 weights, bf16 out.
// transposeOut==0 -> Y is [B*S][H] ; transposeOut==1 -> Y is [B][H][S].
// One wave per 16-row tile; 8 N-tiles x 4 K-steps of v_wmma bf16.
// ---------------------------------------------------------------------------
__global__ __launch_bounds__(32) void proj_kernel(
    const float* __restrict__ X, const bf16* __restrict__ W,
    const float* __restrict__ bias, bf16* __restrict__ Y, int transposeOut) {
  const int lane = threadIdx.x;
  const int half = lane >> 4;
  const int ln   = lane & 15;
  const int row0 = blockIdx.x * 16;  // row in [0, B*S)

  // A fragments (ISA 16-bit A 16x32 layout): lane row = ln;
  // elems 0..7 -> K = half*8+0..7, elems 8..15 -> K = 16+half*8+0..7.
  v16bf a[4];
  const float* xrow = X + (size_t)(row0 + ln) * EE;
  __builtin_prefetch(xrow + 16 * EE, 0, 3);  // next tile (speculative)
#pragma unroll
  for (int kk = 0; kk < 4; ++kk) {
    const int e0 = kk * 32 + half * 8;
#pragma unroll
    for (int i = 0; i < 8; ++i) {
      a[kk][i]     = (bf16)xrow[e0 + i];
      a[kk][8 + i] = (bf16)xrow[e0 + 16 + i];
    }
  }

#pragma unroll
  for (int n = 0; n < 8; ++n) {
    v8f acc;
#pragma unroll
    for (int j = 0; j < 8; ++j) acc[j] = 0.0f;
    const int h = n * 16 + ln;
#pragma unroll
    for (int kk = 0; kk < 4; ++kk) {
      // B fragment: B[e][h] = W[h][e]; lane col = h, K rows = kk*32+half*8+i, kk*32+16+half*8+i
      const bf16* wrow = W + (size_t)h * EE + kk * 32 + half * 8;
      v16bf bfrag = cat8(*(const v8bf*)wrow, *(const v8bf*)(wrow + 16));
      acc = __builtin_amdgcn_wmma_f32_16x16x32_bf16(
          false, a[kk], false, bfrag, (short)0, acc, false, false);
    }
    const float bv = bias[h];
    if (!transposeOut) {
#pragma unroll
      for (int j = 0; j < 8; ++j)
        Y[(size_t)(row0 + j + 8 * half) * HH + h] = (bf16)(acc[j] + bv);
    } else {
      const int b  = row0 / SS;
      const int s0 = row0 % SS;
#pragma unroll
      for (int j = 0; j < 8; ++j)
        Y[((size_t)b * HH + h) * SS + s0 + j + 8 * half] = (bf16)(acc[j] + bv);
    }
  }
}

// ---------------------------------------------------------------------------
// Flash attention: one wave per 16-query tile, 32-key blocks, causal.
// K/V tiles are double-buffered in LDS via gfx1250 async copies
// (global_load_async_to_lds_b128 + s_wait_asynccnt), overlapping the next
// block's data movement with the current block's WMMA + softmax.
// qh,kh: [B][S][H] bf16 ; vhT: [B][H][S] bf16 ; out: [B][S][H] fp32.
// ---------------------------------------------------------------------------
__global__ __launch_bounds__(32) void attn_kernel(
    const bf16* __restrict__ qh, const bf16* __restrict__ kh,
    const bf16* __restrict__ vhT, float* __restrict__ out) {
  __shared__ __align__(16) bf16 kbuf[2][32 * 128];  // 2 x 8KB  K tiles
  __shared__ __align__(16) bf16 vbuf[2][128 * 32];  // 2 x 8KB  V^T tiles
  __shared__ __align__(16) bf16 pbuf[16 * 32];      // 1KB P bounce (C->A)

  const int lane = threadIdx.x;
  const int half = lane >> 4;
  const int ln   = lane & 15;
  const int b    = blockIdx.y;
  const int q0   = blockIdx.x * 16;

  // Issue the async copy of one 32-key block into LDS buffer `buf`.
  auto issue_async = [&](int buf, int k0) {
    // K tile: rows k0..k0+31 are one contiguous 8KB span in [B][S][H];
    // each lane copies 256B (one key row) as 16 b128 async ops.
    {
      unsigned long long ga =
          (unsigned long long)(const void*)(kh + ((size_t)b * SS + k0) * HH) +
          (unsigned long long)lane * 256u;
      unsigned la = (unsigned)(unsigned long long)(const void*)
                        ((const char*)&kbuf[buf][0] + lane * 256);
#pragma unroll
      for (int c = 0; c < 16; ++c)
        asm volatile("global_load_async_to_lds_b128 %0, %1, off offset:%2"
                     :: "v"(la), "v"(ga), "i"(c * 16) : "memory");
    }
    // V^T tile: 128 d-rows x 32 keys; each lane copies 4 rows of 64B.
#pragma unroll
    for (int r = 0; r < 4; ++r) {
      const int d = r * 32 + lane;
      unsigned long long ga = (unsigned long long)(const void*)
          (vhT + ((size_t)b * HH + d) * SS + k0);
      unsigned la = (unsigned)(unsigned long long)(const void*)
                        ((const char*)&vbuf[buf][0] + d * 64);
#pragma unroll
      for (int c = 0; c < 4; ++c)
        asm volatile("global_load_async_to_lds_b128 %0, %1, off offset:%2"
                     :: "v"(la), "v"(ga), "i"(c * 16) : "memory");
    }
  };

  // Q fragments, kept in registers for the whole key loop.
  v16bf qf[4];
  const bf16* qrow = qh + ((size_t)b * SS + q0 + ln) * HH;
#pragma unroll
  for (int kk = 0; kk < 4; ++kk) {
    v8bf lo = *(const v8bf*)(qrow + kk * 32 + half * 8);
    v8bf hi = *(const v8bf*)(qrow + kk * 32 + 16 + half * 8);
    qf[kk] = cat8(lo, hi);
  }

  v8f o[8];
#pragma unroll
  for (int t = 0; t < 8; ++t)
#pragma unroll
    for (int j = 0; j < 8; ++j) o[t][j] = 0.0f;

  float m[8], lsum[8];
#pragma unroll
  for (int j = 0; j < 8; ++j) { m[j] = -__builtin_inff(); lsum[j] = 0.0f; }

  const int kblocks = (q0 + 16 + 31) >> 5;  // causal: keys 0 .. q0+15
  issue_async(0, 0);

  for (int kb = 0; kb < kblocks; ++kb) {
    const int k0  = kb * 32;
    const int buf = kb & 1;

    // Current block's tiles are resident; kick off the next block.
    asm volatile("s_wait_asynccnt 0x0" ::: "memory");
    if (kb + 1 < kblocks) issue_async(buf ^ 1, k0 + 32);

    // ---- S = Q @ K^T : two 16x16 N-tiles, 4 K-steps each (K from LDS) ----
    v8f sacc[2];
#pragma unroll
    for (int n = 0; n < 2; ++n)
#pragma unroll
      for (int j = 0; j < 8; ++j) sacc[n][j] = 0.0f;
#pragma unroll
    for (int n = 0; n < 2; ++n) {
      const bf16* krow = &kbuf[buf][(n * 16 + ln) * 128];
#pragma unroll
      for (int kk = 0; kk < 4; ++kk) {
        v8bf lo = *(const v8bf*)(krow + kk * 32 + half * 8);
        v8bf hi = *(const v8bf*)(krow + kk * 32 + 16 + half * 8);
        v16bf kf = cat8(lo, hi);
        sacc[n] = __builtin_amdgcn_wmma_f32_16x16x32_bf16(
            false, qf[kk], false, kf, (short)0, sacc[n], false, false);
      }
    }

    // ---- mask + scale + online softmax stats ----
    float p0[8], p1[8], mb[8];
#pragma unroll
    for (int j = 0; j < 8; ++j) {
      const int qi = q0 + j + 8 * half;
      float s0 = sacc[0][j] * SCALE;
      float s1 = sacc[1][j] * SCALE;
      if (k0 + ln > qi)      s0 = -3.0e38f;
      if (k0 + 16 + ln > qi) s1 = -3.0e38f;
      p0[j] = s0; p1[j] = s1;
      mb[j] = fmaxf(s0, s1);
    }
#pragma unroll
    for (int off = 1; off < 16; off <<= 1)
#pragma unroll
      for (int j = 0; j < 8; ++j)
        mb[j] = fmaxf(mb[j], __shfl_xor(mb[j], off, 32));

    float alpha[8];
#pragma unroll
    for (int j = 0; j < 8; ++j) {
      const float mn = fmaxf(m[j], mb[j]);
      alpha[j] = __expf(m[j] - mn);
      m[j] = mn;
      p0[j] = __expf(p0[j] - mn);   // masked entries underflow to 0
      p1[j] = __expf(p1[j] - mn);
    }
    float rs[8];
#pragma unroll
    for (int j = 0; j < 8; ++j) rs[j] = p0[j] + p1[j];
#pragma unroll
    for (int off = 1; off < 16; off <<= 1)
#pragma unroll
      for (int j = 0; j < 8; ++j) rs[j] += __shfl_xor(rs[j], off, 32);
#pragma unroll
    for (int j = 0; j < 8; ++j) lsum[j] = lsum[j] * alpha[j] + rs[j];
#pragma unroll
    for (int t = 0; t < 8; ++t)
#pragma unroll
      for (int j = 0; j < 8; ++j) o[t][j] *= alpha[j];

    // ---- P: C layout -> A layout through LDS ----
    asm volatile("" ::: "memory");
#pragma unroll
    for (int j = 0; j < 8; ++j) {
      pbuf[(j + 8 * half) * 32 + ln]      = (bf16)p0[j];
      pbuf[(j + 8 * half) * 32 + 16 + ln] = (bf16)p1[j];
    }
    asm volatile("s_wait_dscnt 0x0" ::: "memory");  // cross-lane visibility
    v16bf pf;
    {
      v8bf lo = *(const v8bf*)(pbuf + ln * 32 + half * 8);
      v8bf hi = *(const v8bf*)(pbuf + ln * 32 + 16 + half * 8);
      pf = cat8(lo, hi);
    }

    // ---- O += P @ V : 8 d-tiles, K=32 keys (V from LDS) ----
#pragma unroll
    for (int t = 0; t < 8; ++t) {
      const bf16* vrow = &vbuf[buf][(t * 16 + ln) * 32 + half * 8];
      v16bf vf = cat8(*(const v8bf*)vrow, *(const v8bf*)(vrow + 16));
      o[t] = __builtin_amdgcn_wmma_f32_16x16x32_bf16(
          false, pf, false, vf, (short)0, o[t], false, false);
    }
  }

  // ---- epilogue: normalize and store fp32 ----
  float invl[8];
#pragma unroll
  for (int j = 0; j < 8; ++j) invl[j] = 1.0f / lsum[j];
#pragma unroll
  for (int t = 0; t < 8; ++t)
#pragma unroll
    for (int j = 0; j < 8; ++j)
      out[((size_t)b * SS + q0 + j + 8 * half) * HH + t * 16 + ln] =
          o[t][j] * invl[j];
}

// ---------------------------------------------------------------------------
extern "C" void kernel_launch(void* const* d_in, const int* in_sizes, int n_in,
                              void* d_out, int out_size, void* d_ws,
                              size_t ws_size, hipStream_t stream) {
  (void)in_sizes; (void)n_in; (void)out_size; (void)ws_size;
  const float* q  = (const float*)d_in[0];
  const float* k  = (const float*)d_in[1];
  const float* v  = (const float*)d_in[2];
  const float* Wq = (const float*)d_in[3];
  const float* bq = (const float*)d_in[4];
  const float* Wk = (const float*)d_in[5];
  const float* bk = (const float*)d_in[6];
  const float* Wv = (const float*)d_in[7];
  const float* bv = (const float*)d_in[8];
  // d_in[9] = mask, unused (causal structure is baked into the kernel)

  const size_t N  = (size_t)BB * SS * HH;  // 4,194,304 elements
  const int    NW = HH * EE;               // 16,384 weight elements
  bf16* qh  = (bf16*)d_ws;        // [B][S][H]
  bf16* kh  = qh + N;             // [B][S][H]
  bf16* vhT = kh + N;             // [B][H][S]
  bf16* Wqb = vhT + N;            // [H][E] bf16
  bf16* Wkb = Wqb + NW;
  bf16* Wvb = Wkb + NW;

  cvt_bf16_kernel<<<dim3(NW / 256), dim3(256), 0, stream>>>(Wq, Wqb, NW);
  cvt_bf16_kernel<<<dim3(NW / 256), dim3(256), 0, stream>>>(Wk, Wkb, NW);
  cvt_bf16_kernel<<<dim3(NW / 256), dim3(256), 0, stream>>>(Wv, Wvb, NW);

  const int tiles = (BB * SS) / 16;  // 2048 row tiles
  proj_kernel<<<dim3(tiles), dim3(32), 0, stream>>>(q, Wqb, bq, qh, 0);
  proj_kernel<<<dim3(tiles), dim3(32), 0, stream>>>(k, Wkb, bk, kh, 0);
  proj_kernel<<<dim3(tiles), dim3(32), 0, stream>>>(v, Wvb, bv, vhT, 1);

  attn_kernel<<<dim3(SS / 16, BB), dim3(32), 0, stream>>>(
      qh, kh, vhT, (float*)d_out);
}
